// RNN_GNN_Fusion_5205500363076
// MI455X (gfx1250) — matrix operand order repaired and checked by reading, not compiled, not run on hardware
//
#include <hip/hip_runtime.h>
#include <hip/hip_bf16.h>

#define HDIM 64
#define TENC 48
#define TF   24

typedef __attribute__((ext_vector_type(16))) __bf16 v16bf;
typedef __attribute__((ext_vector_type(8)))  float  v8f;

__device__ __forceinline__ v8f zero8() {
  v8f c;
  #pragma unroll
  for (int i = 0; i < 8; ++i) c[i] = 0.0f;
  return c;
}

// Fast transcendentals on the hardware pipes: v_exp_f32 + v_rcp_f32.
// sigmoid(x) = rcp(1 + exp(-x)); exp(+inf)->inf -> rcp -> 0 (correct tail).
__device__ __forceinline__ float sigf(float x) {
  return __builtin_amdgcn_rcpf(1.0f + __expf(-x));
}
// tanh(x) = 1 - 2*rcp(exp(2x) + 1); overflow -> 1, underflow -> -1 (correct tails).
__device__ __forceinline__ float tanh_fast(float x) {
  return 1.0f - 2.0f * __builtin_amdgcn_rcpf(__expf(2.0f * x) + 1.0f);
}

// A fragment (16x32, bf16) built from an fp32 LDS tile [16][stride], K window at kb.
// ISA 7.12.2 16-bit A layout: lane group g = lane>>4:
//   elems 0..7  -> K = kb + 8g + i
//   elems 8..15 -> K = kb + 16 + 8g + (i-8)
__device__ __forceinline__ v16bf afrag(const float* sh, int stride, int kb, int lane) {
  const int row = lane & 15;
  const int g8  = (lane >> 4) << 3;
  const float* p = sh + row * stride + kb + g8;
  v16bf r;
  #pragma unroll
  for (int i = 0; i < 8; ++i) r[i] = (__bf16)p[i];
  #pragma unroll
  for (int i = 0; i < 8; ++i) r[8 + i] = (__bf16)p[16 + i];
  return r;
}

// B fragment (32x16, bf16): B[k][n] = W[u0+n][kb+k], W row-major [units][rowlen].
// Same K order as A: elems 0..7 -> K=kb+8g+i, elems 8..15 -> K=kb+16+8g+(i-8).
__device__ __forceinline__ v16bf bfrag(const __bf16* W, int rowlen, int u0, int kb, int lane) {
  const __bf16* p = W + (u0 + (lane & 15)) * rowlen + kb + ((lane >> 4) << 3);
  v16bf r;
  #pragma unroll
  for (int i = 0; i < 8; ++i) { r[i] = p[i]; r[8 + i] = p[16 + i]; }
  return r;
}

__device__ __forceinline__ v8f wmma_bf16(v16bf a, v16bf b, v8f c) {
  return __builtin_amdgcn_wmma_f32_16x16x32_bf16(false, a, false, b, (short)0, c, false, false);
}

// [16x64] x [64x16] tile via two chained K=32 WMMAs
__device__ __forceinline__ v8f gemmK64(v16bf a0, v16bf a1, const __bf16* W, int u0, int lane) {
  v8f c = zero8();
  c = wmma_bf16(a0, bfrag(W, 64, u0,  0, lane), c);
  c = wmma_bf16(a1, bfrag(W, 64, u0, 32, lane), c);
  return c;
}

struct RnnW {
  const float*  wih0;  // [192]   (input dim 1)
  const float*  bih0;  // [192]
  const float*  bhh0;  // [192]
  const __bf16* whh0;  // [192][64]
  const __bf16* wih1;  // [192][64]
  const __bf16* whh1;  // [192][64]
  const float*  bih1;  // [192]
  const float*  bhh1;  // [192]
  const __bf16* out1w; // [64][64]
  const float*  out1b; // [64]
  const float*  out2w; // [64]
};

// One full 2-layer GRU step + output head for a wave-owned 16-node tile.
// h0sh/h1sh: [16*HDIM] fp32 (updated in place). xsh: [16] scalar inputs.
// rs: [16] reduction scratch. Returns head value (pre out2_b) for row `lane` (lanes 0-15).
__device__ float rnn_step(float* h0sh, float* h1sh, const float* xsh, float* rs,
                          const RnnW& W, int lane) {
  const int nl = lane & 15;
  const int mh = (lane >> 4) << 3; // row offset 0 or 8 for C-layout elements

  // ---------------- layer 0: gh0 = h0 @ whh0^T (WMMA), gi0 = rank-1 ----------------
  v16bf a0 = afrag(h0sh, HDIM, 0, lane);
  v16bf a1 = afrag(h0sh, HDIM, 32, lane);
  #pragma unroll
  for (int j = 0; j < 4; ++j) {
    v8f hr = gemmK64(a0, a1, W.whh0,       16 * j, lane);
    v8f hz = gemmK64(a0, a1, W.whh0,  64 + 16 * j, lane);
    v8f hn = gemmK64(a0, a1, W.whh0, 128 + 16 * j, lane);
    const int ur = 16 * j + nl, uz = 64 + 16 * j + nl, un = 128 + 16 * j + nl;
    const float wr = W.wih0[ur], wz = W.wih0[uz], wn = W.wih0[un];
    const float br = W.bih0[ur] + W.bhh0[ur];
    const float bz = W.bih0[uz] + W.bhh0[uz];
    const float bin = W.bih0[un], bhn = W.bhh0[un];
    #pragma unroll
    for (int r = 0; r < 8; ++r) {
      const int m = mh + r;
      const float x  = xsh[m];
      const float h  = h0sh[m * HDIM + 16 * j + nl];
      const float rg = sigf(x * wr + hr[r] + br);
      const float zg = sigf(x * wz + hz[r] + bz);
      const float ng = tanh_fast(x * wn + bin + rg * (hn[r] + bhn));
      h0sh[m * HDIM + 16 * j + nl] = (1.0f - zg) * ng + zg * h; // tile-private element
    }
  }
  __syncthreads();

  // ---------------- layer 1: gi1 = y0 @ wih1^T, gh1 = h1 @ whh1^T ----------------
  v16bf y0a = afrag(h0sh, HDIM, 0, lane);
  v16bf y0b = afrag(h0sh, HDIM, 32, lane);
  v16bf h1a = afrag(h1sh, HDIM, 0, lane);
  v16bf h1b = afrag(h1sh, HDIM, 32, lane);
  #pragma unroll
  for (int j = 0; j < 4; ++j) {
    v8f ir  = gemmK64(y0a, y0b, W.wih1,       16 * j, lane);
    v8f iz  = gemmK64(y0a, y0b, W.wih1,  64 + 16 * j, lane);
    v8f inn = gemmK64(y0a, y0b, W.wih1, 128 + 16 * j, lane);
    v8f hr  = gemmK64(h1a, h1b, W.whh1,       16 * j, lane);
    v8f hz  = gemmK64(h1a, h1b, W.whh1,  64 + 16 * j, lane);
    v8f hn  = gemmK64(h1a, h1b, W.whh1, 128 + 16 * j, lane);
    const int ur = 16 * j + nl, uz = 64 + 16 * j + nl, un = 128 + 16 * j + nl;
    const float br = W.bih1[ur] + W.bhh1[ur];
    const float bz = W.bih1[uz] + W.bhh1[uz];
    const float bin = W.bih1[un], bhn = W.bhh1[un];
    #pragma unroll
    for (int r = 0; r < 8; ++r) {
      const int m = mh + r;
      const float h  = h1sh[m * HDIM + 16 * j + nl];
      const float rg = sigf(ir[r] + hr[r] + br);
      const float zg = sigf(iz[r] + hz[r] + bz);
      const float ng = tanh_fast(inn[r] + bin + rg * (hn[r] + bhn));
      h1sh[m * HDIM + 16 * j + nl] = (1.0f - zg) * ng + zg * h;
    }
  }
  __syncthreads();

  // ---------------- head: relu(y1 @ out1w^T + b1) . out2w ----------------
  if (lane < 16) rs[nl] = 0.0f; // per-wave DS ops are in-order: safe before atomics below
  v16bf ya = afrag(h1sh, HDIM, 0, lane);
  v16bf yb = afrag(h1sh, HDIM, 32, lane);
  float part[8];
  #pragma unroll
  for (int r = 0; r < 8; ++r) part[r] = 0.0f;
  #pragma unroll
  for (int j = 0; j < 4; ++j) {
    v8f acc = gemmK64(ya, yb, W.out1w, 16 * j, lane);
    const int u = 16 * j + nl;
    const float b = W.out1b[u], w2 = W.out2w[u];
    #pragma unroll
    for (int r = 0; r < 8; ++r) {
      float v = acc[r] + b;
      v = v > 0.0f ? v : 0.0f;
      part[r] += v * w2;
    }
  }
  #pragma unroll
  for (int r = 0; r < 8; ++r) atomicAdd(&rs[mh + r], part[r]);
  __syncthreads();
  return (lane < 16) ? rs[nl] : 0.0f;
}

// ---------------------------------------------------------------------------------
// Kernels
// ---------------------------------------------------------------------------------

__global__ void prep_weights(const float* whh0, const float* wih1, const float* whh1,
                             const float* out1w, const float* w1, const float* w2,
                             __bf16* whh0b, __bf16* wih1b, __bf16* whh1b,
                             __bf16* out1wb, __bf16* w1b, __bf16* w2b) {
  const int i = blockIdx.x * blockDim.x + threadIdx.x;
  if (i < 192 * 64) {
    whh0b[i] = (__bf16)whh0[i];
    wih1b[i] = (__bf16)wih1[i];
    whh1b[i] = (__bf16)whh1[i];
  }
  if (i < 64 * 64) out1wb[i] = (__bf16)out1w[i];
  if (i < 64 * 32) {
    const int r = i >> 5, c = i & 31;
    w1b[i] = (c < 14) ? (__bf16)w1[r * 14 + c] : (__bf16)0.0f; // pad K 14 -> 32
  }
  if (i < 32 * 64) w2b[i] = (__bf16)w2[i];
}

__global__ void init_nodes(const float* __restrict__ x, float* seq, float* norm, int n) {
  const int i = blockIdx.x * blockDim.x + threadIdx.x;
  if (i < n) {
    #pragma unroll
    for (int c = 0; c < 6; ++c) seq[i * 32 + c] = x[i * TENC + 42 + c];
    norm[i] = 0.0f;
  }
}

__global__ void norm_accum(const float* __restrict__ ea, const int* __restrict__ dst,
                           float* norm, int ne) {
  const int e = blockIdx.x * blockDim.x + threadIdx.x;
  if (e < ne) atomicAdd(&norm[dst[e]], ea[e]);
}

__global__ void zero_agg(float* agg, int n) {
  const int i = blockIdx.x * blockDim.x + threadIdx.x;
  if (i < n) agg[i] = 0.0f;
}

__global__ __launch_bounds__(128)
void encoder_kernel(const float* __restrict__ x,
                    const float* wih0, const float* bih0, const float* bhh0,
                    const __bf16* whh0, const __bf16* wih1, const __bf16* whh1,
                    const float* bih1, const float* bhh1,
                    const __bf16* out1w, const float* out1b, const float* out2w,
                    const float* out2b,
                    float* enc_out, float* seq, float* h0g, float* h1g, int n) {
  __shared__ float lds[4 * (2 * 16 * HDIM + 32)];
  const int wave = threadIdx.x >> 5;
  const int lane = threadIdx.x & 31;
  float* h0sh = lds + wave * (2 * 16 * HDIM + 32);
  float* h1sh = h0sh + 16 * HDIM;
  float* xsh  = h1sh + 16 * HDIM;
  float* rs   = xsh + 16;
  const int nodeBase = (blockIdx.x * 4 + wave) * 16;

  for (int i = lane; i < 16 * HDIM; i += 32) { h0sh[i] = 0.0f; h1sh[i] = 0.0f; }
  __syncthreads();

  RnnW W = {wih0, bih0, bhh0, whh0, wih1, whh1, bih1, bhh1, out1w, out1b, out2w};
  const float ob = *out2b;

  #pragma unroll 1
  for (int t = 0; t < TENC; ++t) {
    if (lane < 16) {
      const int node = nodeBase + lane;
      xsh[lane] = (node < n) ? x[node * TENC + t] : 0.0f;
    }
    __syncthreads();
    const float y = rnn_step(h0sh, h1sh, xsh, rs, W, lane) + ob;
    if (lane < 16) {
      const int node = nodeBase + lane;
      if (node < n) {
        enc_out[node * TENC + t] = y;
        if (t == TENC - 1) seq[node * 32 + 6] = y; // y_prev seed for decoder
      }
    }
    __syncthreads();
  }
  for (int i = lane; i < 16 * HDIM; i += 32) {
    const int row = i / HDIM, col = i % HDIM;
    const int node = nodeBase + row;
    if (node < n) { h0g[node * HDIM + col] = h0sh[i]; h1g[node * HDIM + col] = h1sh[i]; }
  }
}

__global__ __launch_bounds__(128)
void gnn_kernel(const float* __restrict__ seq, const int* __restrict__ src,
                const int* __restrict__ dst, const float* __restrict__ ea,
                const __bf16* w1b, const float* b1,
                const __bf16* w2b, const float* b2,
                const float* wout, const float* bout,
                float* agg, int ne, int k) {
  __shared__ float lds[4 * (16 * 32 + 16 * HDIM + 16 + 16 + 16)];
  const int wave = threadIdx.x >> 5;
  const int lane = threadIdx.x & 31;
  float* msh  = lds + wave * (16 * 32 + 16 * HDIM + 48);
  float* hid  = msh + 16 * 32;
  float* rs   = hid + 16 * HDIM;
  float* eash = rs + 16;
  int*   dsts = (int*)(eash + 16);
  const int eBase = (blockIdx.x * 4 + wave) * 16;
  const int nl = lane & 15;
  const int mh = (lane >> 4) << 3;

  if (lane < 16) {
    int e = eBase + lane;
    if (e >= ne) e = ne - 1; // clamp loads; store guarded below
    const int sN = src[e], dN = dst[e];
    #pragma unroll
    for (int c = 0; c < 7; ++c) {
      const float xj = seq[sN * 32 + k + c];
      const float xi = seq[dN * 32 + k + c];
      msh[lane * 32 + c]     = xj;
      msh[lane * 32 + 7 + c] = xj - xi;
    }
    #pragma unroll
    for (int c = 14; c < 32; ++c) msh[lane * 32 + c] = 0.0f;
    eash[lane] = ea[e];
    dsts[lane] = dN;
    rs[lane]   = 0.0f;
  }
  __syncthreads();

  // layer 1: [16x14->64], K padded to 32 -> one WMMA per 16-unit tile
  v16bf am = afrag(msh, 32, 0, lane);
  #pragma unroll
  for (int j = 0; j < 4; ++j) {
    v8f acc = wmma_bf16(am, bfrag(w1b, 32, 16 * j, 0, lane), zero8());
    const int u = 16 * j + nl;
    const float b = b1[u];
    #pragma unroll
    for (int r = 0; r < 8; ++r) {
      float v = acc[r] + b;
      hid[(mh + r) * HDIM + 16 * j + nl] = v > 0.0f ? v : 0.0f;
    }
  }
  __syncthreads();

  // layer 2: [16x64->32] + layer 3 dot(wout) fused
  v16bf ha = afrag(hid, HDIM, 0, lane);
  v16bf hb = afrag(hid, HDIM, 32, lane);
  float part[8];
  #pragma unroll
  for (int r = 0; r < 8; ++r) part[r] = 0.0f;
  #pragma unroll
  for (int j = 0; j < 2; ++j) {
    v8f acc = gemmK64(ha, hb, w2b, 16 * j, lane);
    const int u = 16 * j + nl;
    const float b = b2[u], w = wout[u];
    #pragma unroll
    for (int r = 0; r < 8; ++r) {
      float v = acc[r] + b;
      v = v > 0.0f ? v : 0.0f;
      part[r] += v * w;
    }
  }
  #pragma unroll
  for (int r = 0; r < 8; ++r) atomicAdd(&rs[mh + r], part[r]);
  __syncthreads();

  if (lane < 16 && (eBase + lane) < ne) {
    const float m = (rs[lane] + *bout) * eash[lane];
    atomicAdd(&agg[dsts[lane]], m); // L2-resident f32 atomic segment-sum
  }
}

__global__ __launch_bounds__(128)
void dec_rnn_kernel(const float* wih0, const float* bih0, const float* bhh0,
                    const __bf16* whh0, const __bf16* wih1, const __bf16* whh1,
                    const float* bih1, const float* bhh1,
                    const __bf16* out1w, const float* out1b, const float* out2w,
                    const float* out2b, const float* out_param, const float* lin_w,
                    const float* __restrict__ agg, const float* __restrict__ norm,
                    float* seq, float* dec_out, float* h0g, float* h1g, int n, int k) {
  __shared__ float lds[4 * (2 * 16 * HDIM + 32)];
  const int wave = threadIdx.x >> 5;
  const int lane = threadIdx.x & 31;
  float* h0sh = lds + wave * (2 * 16 * HDIM + 32);
  float* h1sh = h0sh + 16 * HDIM;
  float* xsh  = h1sh + 16 * HDIM;
  float* rs   = xsh + 16;
  const int nodeBase = (blockIdx.x * 4 + wave) * 16;

  for (int i = lane; i < 16 * HDIM; i += 32) {
    const int row = i / HDIM, col = i % HDIM;
    const int node = nodeBase + row;
    h0sh[i] = (node < n) ? h0g[node * HDIM + col] : 0.0f;
    h1sh[i] = (node < n) ? h1g[node * HDIM + col] : 0.0f;
  }
  if (lane < 16) {
    const int node = nodeBase + lane;
    xsh[lane] = (node < n) ? seq[node * 32 + 6 + k] : 0.0f;
  }
  __syncthreads();

  RnnW W = {wih0, bih0, bhh0, whh0, wih1, whh1, bih1, bhh1, out1w, out1b, out2w};
  const float y = rnn_step(h0sh, h1sh, xsh, rs, W, lane) + *out2b;

  if (lane < 16) {
    const int node = nodeBase + lane;
    if (node < n) {
      const float s    = sigf(*out_param);
      const float diff = (agg[node] / norm[node]) * (*lin_w);
      const float out  = s * y + (1.0f - s) * diff;
      dec_out[node * TF + k] = out;
      seq[node * 32 + 7 + k] = out; // feed next decoder step's x7 window
    }
  }
  for (int i = lane; i < 16 * HDIM; i += 32) {
    const int row = i / HDIM, col = i % HDIM;
    const int node = nodeBase + row;
    if (node < n) { h0g[node * HDIM + col] = h0sh[i]; h1g[node * HDIM + col] = h1sh[i]; }
  }
}

// ---------------------------------------------------------------------------------
extern "C" void kernel_launch(void* const* d_in, const int* in_sizes, int n_in,
                              void* d_out, int out_size, void* d_ws, size_t ws_size,
                              hipStream_t stream) {
  const float* x        = (const float*)d_in[0];   // [N,48,1]
  const float* ea       = (const float*)d_in[1];   // [E]
  const int*   eidx     = (const int*)d_in[2];     // [2,E]
  const float* wih0     = (const float*)d_in[3];
  const float* whh0f    = (const float*)d_in[4];
  const float* bih0     = (const float*)d_in[5];
  const float* bhh0     = (const float*)d_in[6];
  const float* wih1f    = (const float*)d_in[7];
  const float* whh1f    = (const float*)d_in[8];
  const float* bih1     = (const float*)d_in[9];
  const float* bhh1     = (const float*)d_in[10];
  const float* out1wf   = (const float*)d_in[11];
  const float* out1b    = (const float*)d_in[12];
  const float* out2w    = (const float*)d_in[13];
  const float* out2b    = (const float*)d_in[14];
  const float* w1f      = (const float*)d_in[15];
  const float* b1       = (const float*)d_in[16];
  const float* w2f      = (const float*)d_in[17];
  const float* b2       = (const float*)d_in[18];
  const float* wout     = (const float*)d_in[19];
  const float* bout     = (const float*)d_in[20];
  const float* lin_w    = (const float*)d_in[21];
  const float* out_prm  = (const float*)d_in[22];

  const int N = in_sizes[0] / TENC;
  const int E = in_sizes[1];
  const int* src = eidx;
  const int* dst = eidx + E;

  float* dec_out = (float*)d_out;                  // [N,24]
  float* enc_out = dec_out + (size_t)N * TF;       // [N,48]

  // workspace layout
  float* h0   = (float*)d_ws;
  float* h1   = h0 + (size_t)N * HDIM;
  float* seq  = h1 + (size_t)N * HDIM;             // [N,32] rolling history
  float* agg  = seq + (size_t)N * 32;
  float* norm = agg + N;
  __bf16* whh0b  = (__bf16*)(norm + N);
  __bf16* wih1b  = whh0b + 192 * 64;
  __bf16* whh1b  = wih1b + 192 * 64;
  __bf16* out1wb = whh1b + 192 * 64;
  __bf16* w1b    = out1wb + 64 * 64;               // [64][32] zero-padded
  __bf16* w2b    = w1b + 64 * 32;

  prep_weights<<<(192 * 64 + 255) / 256, 256, 0, stream>>>(
      whh0f, wih1f, whh1f, out1wf, w1f, w2f, whh0b, wih1b, whh1b, out1wb, w1b, w2b);
  init_nodes<<<(N + 255) / 256, 256, 0, stream>>>(x, seq, norm, N);
  norm_accum<<<(E + 255) / 256, 256, 0, stream>>>(ea, dst, norm, E);

  const int nbRnn = (N + 63) / 64;
  encoder_kernel<<<nbRnn, 128, 0, stream>>>(
      x, wih0, bih0, bhh0, whh0b, wih1b, whh1b, bih1, bhh1,
      out1wb, out1b, out2w, out2b, enc_out, seq, h0, h1, N);

  const int nbG = (E + 63) / 64;
  for (int k = 0; k < TF; ++k) {
    zero_agg<<<(N + 255) / 256, 256, 0, stream>>>(agg, N);
    gnn_kernel<<<nbG, 128, 0, stream>>>(seq, src, dst, ea, w1b, b1, w2b, b2,
                                        wout, bout, agg, E, k);
    dec_rnn_kernel<<<nbRnn, 128, 0, stream>>>(
        wih0, bih0, bhh0, whh0b, wih1b, whh1b, bih1, bhh1,
        out1wb, out1b, out2w, out2b, out_prm, lin_w,
        agg, norm, seq, dec_out, h0, h1, N, k);
  }
}
